// MaskedEmbedding_74036646248514
// MI455X (gfx1250) — matrix operand order repaired and checked by reading, not compiled, not run on hardware
//
#include <hip/hip_runtime.h>
#include <cstddef>

// Problem constants (match reference)
constexpr int Bsz = 256, Esz = 64, Gsz = 2000, Psz = 512;
// Tiling
constexpr int BT = 128;   // output rows (b) per workgroup
constexpr int PT = 128;   // output cols (p) per workgroup
constexpr int KC = 32;    // k (g) chunk per iteration
constexpr int LDW = 20;   // LDS row stride in dwords (16 payload + 4 pad; multiple of 4 keeps b128 alignment)
constexpr int NCH = (Gsz + KC - 1) / KC;  // 63 k-chunks (last one ragged: 16 valid g)

typedef __attribute__((ext_vector_type(16))) __bf16 bf16x16;
typedef __attribute__((ext_vector_type(8)))  float  f32x8;
typedef __attribute__((ext_vector_type(4)))  float  float4v;
typedef __attribute__((ext_vector_type(4)))  unsigned uint4v;
typedef __attribute__((ext_vector_type(2)))  unsigned uint2v;

union Frag {
    uint4v  q[2];
    bf16x16 v;
};

// float pair -> packed 2x bf16 via the hardware converter (round-to-nearest-even).
// CDNA5 bf16 VALU: v_cvt_pk_bf16_f32 dst.lo = bf16(src0), dst.hi = bf16(src1)
__device__ __forceinline__ unsigned pack_bf16(float a, float b) {
    unsigned r;
    asm("v_cvt_pk_bf16_f32 %0, %1, %2" : "=v"(r) : "v"(a), "v"(b));
    return r;
}

// ---- staging loads: uniform SGPR base + one 32-bit VGPR offset + immediate offsets ----
// X tile: 128 rows x 32 k. Thread's 4 tasks differ by 32 rows -> immediate 32*Gsz elements.
__device__ __forceinline__ void load_x(const float* __restrict__ Xblk,  // X + bBase*Gsz (uniform)
                                       int g0, int tid, float4v q[4]) {
    int b0 = tid >> 3;                       // row 0..31 for i=0
    int kq = tid & 7;                        // k-quad 0..7 (same for all i)
    int g  = g0 + 4 * kq;
    int gc = g < Gsz ? g : (Gsz - 4);        // quad clamp: in-bounds, 16B aligned
    int voff = b0 * Gsz + gc;                // 32-bit element offset
    #pragma unroll
    for (int i = 0; i < 4; ++i)
        q[i] = *(const float4v*)(Xblk + voff + i * 32 * Gsz);
}

// W & mask tile: 32 k-rows x 128 p. Thread: 4 consecutive (clamped-quad) g-rows x float4 p.
__device__ __forceinline__ void load_w(const float* __restrict__ Wblk,  // W + e*G*P + pBase (uniform)
                                       const float* __restrict__ Mblk,  // Mask + pBase (uniform)
                                       int g0, int tid,
                                       float4v wq[4], float4v mq[4]) {
    int pg = tid & 31;                       // p-group (consecutive lanes -> coalesced)
    int gq = tid >> 5;                       // g-quad 0..7 (one per wave)
    int g  = g0 + 4 * gq;
    int gb = g < Gsz ? g : (Gsz - 4);        // whole-quad clamp: rows gb..gb+3 in-bounds
    int voff = gb * Psz + 4 * pg;            // 32-bit element offset
    #pragma unroll
    for (int r = 0; r < 4; ++r) {
        wq[r] = *(const float4v*)(Wblk + voff + r * Psz);
        mq[r] = *(const float4v*)(Mblk + voff + r * Psz);
    }
}

// ---- staging stores: convert + pack + b64 LDS store ----
__device__ __forceinline__ void store_x(unsigned* __restrict__ sXb, int g0, int tid,
                                        const float4v q[4]) {
    int b0 = tid >> 3;
    int kq = tid & 7;
    bool valid = (g0 + 4 * kq) < Gsz;        // uniform per thread across i (Gsz % 4 == 0)
    #pragma unroll
    for (int i = 0; i < 4; ++i) {
        uint2v u;
        u[0] = pack_bf16(q[i][0], q[i][1]);
        u[1] = pack_bf16(q[i][2], q[i][3]);
        u[0] = valid ? u[0] : 0u;            // zero X side only; 0 * finite(W) = 0 in WMMA
        u[1] = valid ? u[1] : 0u;
        *(uint2v*)&sXb[(b0 + i * 32) * LDW + 2 * kq] = u;  // even dword -> 8B aligned
    }
}

__device__ __forceinline__ void store_w(unsigned* __restrict__ sWb, int tid,
                                        const float4v wq[4], const float4v mq[4]) {
    int pg = tid & 31;
    int gq = tid >> 5;
    #pragma unroll
    for (int j = 0; j < 4; ++j) {            // j = p within quad; transpose into sW[p][k]
        uint2v u;
        u[0] = pack_bf16(wq[0][j] * mq[0][j], wq[1][j] * mq[1][j]);
        u[1] = pack_bf16(wq[2][j] * mq[2][j], wq[3][j] * mq[3][j]);
        *(uint2v*)&sWb[(4 * pg + j) * LDW + 2 * gq] = u;
    }
}

__global__ __launch_bounds__(256)
void masked_expert_gemm(const float* __restrict__ X,     // [B,G]
                        const float* __restrict__ W,     // [E,G,P]
                        const float* __restrict__ Bias,  // [E,P]
                        const float* __restrict__ Mask,  // [G,P]
                        float* __restrict__ Out)         // [B,E,P]
{
    // double-buffered LDS
    __shared__ __attribute__((aligned(16))) unsigned sX[2][BT * LDW];
    __shared__ __attribute__((aligned(16))) unsigned sW[2][PT * LDW];

    const int tid   = threadIdx.x;
    const int e     = blockIdx.z;
    const int pBase = blockIdx.x * PT;
    const int bBase = blockIdx.y * BT;

    const int lane   = tid & 31;
    const int wid    = tid >> 5;
    const int wm     = wid >> 2;      // 0..1 : wave row (64 rows each)
    const int wn     = wid & 3;       // 0..3 : wave col (32 cols each)
    const int laneM  = lane & 15;
    const int laneHi = lane >> 4;     // 0 or 1

    f32x8 acc[4][2] = {};             // 4 M-subtiles x 2 N-subtiles of 16x16 f32

    // uniform (SGPR) base pointers; per-lane offsets stay 32-bit
    const float* Xblk = X + (size_t)bBase * Gsz;
    const float* Wblk = W + (size_t)e * Gsz * Psz + pBase;
    const float* Mblk = Mask + pBase;

    // precomputed fragment LDS offsets (dwords)
    const int offB0 = (wn * 32 + 0 * 16 + laneM) * LDW + laneHi * 8;
    const int offB1 = (wn * 32 + 1 * 16 + laneM) * LDW + laneHi * 8;
    const int offA  = (wm * 64 + laneM) * LDW + laneHi * 8;  // + mt*16*LDW per subtile

    // -------- software pipeline: prologue stages chunk 0 into buffer 0 --------
    float4v xq[4], wq[4], mq[4];
    load_x(Xblk, 0, tid, xq);
    load_w(Wblk, Mblk, 0, tid, wq, mq);
    store_x(sX[0], 0, tid, xq);
    store_w(sW[0], tid, wq, mq);

    for (int c = 0; c < NCH; ++c) {
        __syncthreads();  // buffer c&1 fully staged; buffer (c+1)&1 free

        const bool more = (c + 1) < NCH;
        const int gn = (c + 1) * KC;
        if (more) {  // issue next chunk's global loads early (latency hidden by WMMAs)
            load_x(Xblk, gn, tid, xq);
            load_w(Wblk, Mblk, gn, tid, wq, mq);
        }

        const unsigned* sXb = sX[c & 1];
        const unsigned* sWb = sW[c & 1];

        // B fragments: 32x16 bf16; lane = N col, K = idx + laneHi*16 (contiguous run)
        Frag fb[2];
        fb[0].q[0] = *(const uint4v*)&sWb[offB0];
        fb[0].q[1] = *(const uint4v*)&sWb[offB0 + 4];
        fb[1].q[0] = *(const uint4v*)&sWb[offB1];
        fb[1].q[1] = *(const uint4v*)&sWb[offB1 + 4];

        // A fragments + WMMA. A 16x32 bf16: lane = M, K = idx + laneHi*16 (contiguous run)
        #pragma unroll
        for (int mt = 0; mt < 4; ++mt) {
            Frag fa;
            int off = offA + mt * 16 * LDW;
            fa.q[0] = *(const uint4v*)&sXb[off];       // K = laneHi*16 .. +7
            fa.q[1] = *(const uint4v*)&sXb[off + 4];   // K = laneHi*16+8 .. +7
            #pragma unroll
            for (int nt = 0; nt < 2; ++nt) {
                acc[mt][nt] = __builtin_amdgcn_wmma_f32_16x16x32_bf16(
                    false, fa.v, false, fb[nt].v,
                    (short)0, acc[mt][nt], false, false);
            }
        }

        if (more) {  // convert + stage into the other buffer (disjoint from buffer in use)
            store_x(sX[(c + 1) & 1], gn, tid, xq);
            store_w(sW[(c + 1) & 1], tid, wq, mq);
        }
    }

    // ---- epilogue: add bias, store f32. C layout: VGPR r -> M = r + 8*laneHi, N = laneM ----
    #pragma unroll
    for (int mt = 0; mt < 4; ++mt) {
        #pragma unroll
        for (int nt = 0; nt < 2; ++nt) {
            int col = pBase + wn * 32 + nt * 16 + laneM;
            float bv = Bias[(size_t)e * Psz + col];
            int rowBase = bBase + wm * 64 + mt * 16 + laneHi * 8;
            #pragma unroll
            for (int r = 0; r < 8; ++r) {
                int row = rowBase + r;
                Out[((size_t)row * Esz + e) * Psz + col] = acc[mt][nt][r] + bv;
            }
        }
    }
}

extern "C" void kernel_launch(void* const* d_in, const int* in_sizes, int n_in,
                              void* d_out, int out_size, void* d_ws, size_t ws_size,
                              hipStream_t stream) {
    const float* x    = (const float*)d_in[0];  // [B,G]
    const float* w    = (const float*)d_in[1];  // [E,G,P]
    const float* bias = (const float*)d_in[2];  // [E,P]
    const float* mask = (const float*)d_in[3];  // [G,P]
    float* out = (float*)d_out;                 // [B,E,P]

    dim3 grid(Psz / PT, Bsz / BT, Esz);  // (4, 2, 64)
    dim3 block(256);
    masked_expert_gemm<<<grid, block, 0, stream>>>(x, w, bias, mask, out);
    (void)in_sizes; (void)n_in; (void)out_size; (void)d_ws; (void)ws_size;
}
